// BiLevelRoutingAttention_73409581023486
// MI455X (gfx1250) — matrix operand order repaired and checked by reading, not compiled, not run on hardware
//
#include <hip/hip_runtime.h>
#include <hip/hip_bf16.h>

// ---------- plain ext-vector types (trivially unionable, unlike HIP uint4) ----
typedef __attribute__((ext_vector_type(16))) __bf16        bf16x16;
typedef __attribute__((ext_vector_type(8)))  float         f32x8;
typedef __attribute__((ext_vector_type(4)))  float         f32x4;
typedef __attribute__((ext_vector_type(4)))  unsigned int  u32x4;
typedef int v4i_t __attribute__((__vector_size__(16)));   // matches async-LDS builtin param

union FragAB { u32x4 u[2]; bf16x16 v; };

__device__ __forceinline__ unsigned short f2bf(float f) {
  unsigned int u = __float_as_uint(f);
  u += 0x7fffu + ((u >> 16) & 1u);          // round-to-nearest-even
  return (unsigned short)(u >> 16);
}
__device__ __forceinline__ float bf2f(unsigned short s) {
  return __uint_as_float(((unsigned int)s) << 16);
}

// Problem constants (shapes fixed by setup_inputs)
#define NTOK   65536      // B*T*S
#define CDIM   256
#define HEADS  8
#define DHEAD  32
#define NWIN   8
#define WIN    32
#define TK     4
#define SCALE  0.17677669529663687f   // 32^-0.5

#if defined(__has_builtin)
#if __has_builtin(__builtin_amdgcn_global_load_async_to_lds_b128)
#define HAVE_ASYNC_LDS 1
#endif
#endif

__device__ __forceinline__ void wait_asynccnt0() {
#if defined(__has_builtin) && __has_builtin(__builtin_amdgcn_s_wait_asynccnt)
  __builtin_amdgcn_s_wait_asynccnt(0);
#else
  asm volatile("s_wait_asynccnt 0x0" ::: "memory");
#endif
}

// =====================================================================
// GEMM: C[M,N] = A[M,K] @ W[K,N] + bias  (bf16 WMMA, fp32 accumulate)
// BM=128, BN=128, BK=32; 256 threads = 8 waves; wave w owns M-subtile w
// and all 8 N-subtiles (8 WMMAs per K-step per wave).
// =====================================================================
template<bool A_IS_BF16, bool C_IS_BF16>
__global__ __launch_bounds__(256)
void gemm_bias_kernel(const void* __restrict__ A_,
                      const float* __restrict__ W,
                      const float* __restrict__ bias,
                      void* __restrict__ C_,
                      int N, int K)
{
  constexpr int BM = 128, BN = 128, BK = 32, PAD = 8;
  __shared__ __align__(16) unsigned short As[BM][BK + PAD];   // bf16 bits
  __shared__ __align__(16) unsigned short Ws[BN][BK + PAD];   // W^T, bf16 bits

  const int tid  = threadIdx.x;
  const int lane = tid & 31;
  const int wave = tid >> 5;
  const int ln   = lane & 15;
  const int lh   = lane >> 4;
  const int m0   = blockIdx.x * BM;
  const int n0   = blockIdx.y * BN;

  f32x8 acc[8];
#pragma unroll
  for (int i = 0; i < 8; ++i) acc[i] = (f32x8){0,0,0,0,0,0,0,0};

  const int arow = tid >> 1;            // 0..127
  const int acol = (tid & 1) * 16;      // 0 | 16
  const int wrow = tid >> 3;            // 0..31 (k)
  const int wcol = (tid & 7) * 16;      // n within tile (0..112 step 16)

  for (int kb = 0; kb < K; kb += BK) {
    // ---- stage A tile (convert fp32 -> bf16 if needed) ----
    if constexpr (A_IS_BF16) {
      const unsigned short* Ab  = (const unsigned short*)A_;
      const unsigned short* src = Ab + (size_t)(m0 + arow) * K + kb + acol;
      unsigned short*       dst = &As[arow][acol];
#ifdef HAVE_ASYNC_LDS
      // 32B per thread, ASYNCcnt-tracked DMA into LDS (no VGPR round trip)
      __builtin_amdgcn_global_load_async_to_lds_b128(
          (v4i_t*)src, (v4i_t*)dst, 0, 0);
      __builtin_amdgcn_global_load_async_to_lds_b128(
          (v4i_t*)(src + 8), (v4i_t*)(dst + 8), 0, 0);
#else
      const u32x4* s4 = (const u32x4*)src;
      u32x4*       d4 = (u32x4*)dst;
      d4[0] = s4[0]; d4[1] = s4[1];
#endif
      if (kb + BK < K) __builtin_prefetch(src + BK, 0, 0);
    } else {
      const float* Af = (const float*)A_;
      const float* srcf = Af + (size_t)(m0 + arow) * K + kb + acol;
      const f32x4* src = (const f32x4*)srcf;
      unsigned int* dst = (unsigned int*)&As[arow][acol];
#pragma unroll
      for (int i = 0; i < 4; ++i) {
        f32x4 f = src[i];
        dst[2*i+0] = (unsigned)f2bf(f.x) | ((unsigned)f2bf(f.y) << 16);
        dst[2*i+1] = (unsigned)f2bf(f.z) | ((unsigned)f2bf(f.w) << 16);
      }
      if (kb + BK < K) __builtin_prefetch(srcf + BK, 0, 0);
    }
    // ---- stage W tile transposed: Ws[n][k] ----
    {
      const float* srcf = W + (size_t)(kb + wrow) * N + n0 + wcol;
      const f32x4* src = (const f32x4*)srcf;
#pragma unroll
      for (int i = 0; i < 4; ++i) {
        f32x4 f = src[i];
        Ws[wcol + 4*i + 0][wrow] = f2bf(f.x);
        Ws[wcol + 4*i + 1][wrow] = f2bf(f.y);
        Ws[wcol + 4*i + 2][wrow] = f2bf(f.z);
        Ws[wcol + 4*i + 3][wrow] = f2bf(f.w);
      }
      if (kb + BK < K) __builtin_prefetch(srcf + (size_t)BK * N, 0, 0);
    }
#ifdef HAVE_ASYNC_LDS
    if constexpr (A_IS_BF16) wait_asynccnt0();
#endif
    __syncthreads();

    // A fragment: row = wave*16+ln; K segments {8*lh..}, {16+8*lh..}  (ISA A layout)
    FragAB a;
    a.u[0] = *(const u32x4*)&As[wave*16 + ln][8*lh];
    a.u[1] = *(const u32x4*)&As[wave*16 + ln][16 + 8*lh];

#pragma unroll
    for (int ns = 0; ns < 8; ++ns) {
      // B fragment: col = ns*16+ln; K segments {8*lh..}, {16+8*lh..}  (same as A)
      FragAB b;
      b.u[0] = *(const u32x4*)&Ws[ns*16 + ln][8*lh];
      b.u[1] = *(const u32x4*)&Ws[ns*16 + ln][16 + 8*lh];
      acc[ns] = __builtin_amdgcn_wmma_f32_16x16x32_bf16(
          false, a.v, false, b.v, (short)0, acc[ns], false, false);
    }
    __syncthreads();
  }

  // ---- epilogue: D layout row = 8*lh + v, col = ln ----
#pragma unroll
  for (int ns = 0; ns < 8; ++ns) {
    const int col = n0 + ns*16 + ln;
    const float bv = bias[col];
#pragma unroll
    for (int v = 0; v < 8; ++v) {
      const size_t row = (size_t)(m0 + wave*16 + 8*lh + v);
      const float val = acc[ns][v] + bv;
      if constexpr (C_IS_BF16)
        ((unsigned short*)C_)[row * N + col] = f2bf(val);
      else
        ((float*)C_)[row * N + col] = val;
    }
  }
}

// =====================================================================
// Routing: one wave per (bt, h). Region sums, 8x8 sim, activity mask,
// top-4 region indices per query window.
// =====================================================================
__global__ __launch_bounds__(32)
void routing_kernel(const unsigned short* __restrict__ qkv, int* __restrict__ gidx)
{
  __shared__ float Q[NWIN][DHEAD], Kr[NWIN][DHEAD], Ap[NWIN][DHEAD], Sim[NWIN][NWIN];
  const int u    = blockIdx.x;        // bt*8 + h
  const int h    = u & 7;
  const int bt   = u >> 3;
  const int lane = threadIdx.x;       // d index 0..31
  const size_t base = (size_t)bt * 256 * 768 + (size_t)h * DHEAD + lane;

#pragma unroll
  for (int n = 0; n < NWIN; ++n) {
    float qs = 0.f, ks = 0.f, ap = 0.f;
    for (int t = 0; t < WIN; ++t) {
      const size_t off = base + (size_t)(n*WIN + t) * 768;
      const float qv = bf2f(qkv[off]);
      const float kv = bf2f(qkv[off + 256]);
      qs += qv; ks += kv; ap += fabsf(kv);
    }
    Q[n][lane] = qs; Kr[n][lane] = ks; Ap[n][lane] = ap;
  }
  __syncthreads();  // single-wave workgroup: barrier is a NOP, DS is in-order

#pragma unroll
  for (int j = 0; j < 2; ++j) {
    const int idx = lane * 2 + j;
    const int n = idx >> 3, m = idx & 7;
    float dot = 0.f, act = 0.f;
#pragma unroll
    for (int d2 = 0; d2 < DHEAD; ++d2) { dot += Q[n][d2] * Kr[m][d2]; act += Ap[m][d2]; }
    Sim[n][m] = dot * SCALE + (act > 1e-5f ? 0.f : -1e9f);
  }
  __syncthreads();

  if (lane < NWIN) {
    float row[NWIN];
#pragma unroll
    for (int m = 0; m < NWIN; ++m) row[m] = Sim[lane][m];
    int* out = gidx + ((size_t)u * NWIN + lane) * TK;
#pragma unroll
    for (int t = 0; t < TK; ++t) {
      float best = -3.4e38f; int bj = 0;
#pragma unroll
      for (int m = 0; m < NWIN; ++m) if (row[m] > best) { best = row[m]; bj = m; }
      out[t] = bj;
      row[bj] = -3.4e38f;
    }
  }
}

// =====================================================================
// Windowed attention: one wave per (bt, h, window). 128 gathered keys.
//   S = q @ Kg^T (16 WMMA), row softmax (shfl within 16-lane half),
//   O = P @ Vg   (16 WMMA, P and V^T staged in LDS as bf16).
// =====================================================================
__global__ __launch_bounds__(128)
void attn_kernel(const unsigned short* __restrict__ qkv,
                 const int* __restrict__ gidx,
                 unsigned short* __restrict__ attnb)
{
  __shared__ __align__(16) unsigned short Pbuf[4][32][128];  // per-wave P (bf16)
  __shared__ __align__(16) unsigned short VT[4][32][128];    // per-wave V^T (bf16)

  const int lane = threadIdx.x & 31;
  const int w    = threadIdx.x >> 5;
  const int ln   = lane & 15;
  const int lh   = lane >> 4;
  const int u    = blockIdx.x * 4 + w;   // ((bt)*8+h)*8+n
  const int n    = u & 7;
  const int h    = (u >> 3) & 7;
  const int bt   = u >> 6;
  const size_t tokbase = (size_t)bt * 256;

  int idx4[TK];
#pragma unroll
  for (int i = 0; i < TK; ++i) idx4[i] = gidx[(size_t)u * TK + i];

  // ---- stage gathered V transposed: VT[d][g] ----
#pragma unroll
  for (int i = 0; i < 4; ++i) {
    const int g = i * 32 + lane;
    const int s = idx4[i] * WIN + lane;
    const unsigned short* vp = qkv + (tokbase + s) * 768 + 512 + h * DHEAD;
    union { u32x4 q[4]; unsigned short sh[32]; } vv;
    const u32x4* p = (const u32x4*)vp;
    vv.q[0] = p[0]; vv.q[1] = p[1]; vv.q[2] = p[2]; vv.q[3] = p[3];
#pragma unroll
    for (int d2 = 0; d2 < DHEAD; ++d2) VT[w][d2][g] = vv.sh[d2];
  }

  // ---- S = q Kg^T, softmax, P -> LDS ----
#pragma unroll
  for (int mt = 0; mt < 2; ++mt) {
    const int srow = n * WIN + mt * 16 + ln;
    const unsigned short* qp = qkv + (tokbase + srow) * 768 + h * DHEAD;
    FragAB a;
    a.u[0] = *(const u32x4*)(qp + 8 * lh);
    a.u[1] = *(const u32x4*)(qp + 16 + 8 * lh);

    f32x8 sfr[8];
#pragma unroll
    for (int nt = 0; nt < 8; ++nt) {
      const int skey = idx4[nt >> 1] * WIN + (nt & 1) * 16 + ln;
      const unsigned short* kp = qkv + (tokbase + skey) * 768 + 256 + h * DHEAD;
      FragAB b;
      b.u[0] = ((const u32x4*)kp)[lh];
      b.u[1] = ((const u32x4*)kp)[2 + lh];
      const f32x8 z = (f32x8){0,0,0,0,0,0,0,0};
      sfr[nt] = __builtin_amdgcn_wmma_f32_16x16x32_bf16(
          false, a.v, false, b.v, (short)0, z, false, false);
    }
    // lane holds rows {mt*16 + 8*lh + v}, cols {nt*16 + ln}; row spans the
    // 16 lanes of this half -> xor-reduce with masks 8,4,2,1.
#pragma unroll
    for (int v = 0; v < 8; ++v) {
      float mx = -3.4e38f;
#pragma unroll
      for (int nt = 0; nt < 8; ++nt) mx = fmaxf(mx, sfr[nt][v]);
#pragma unroll
      for (int off = 8; off >= 1; off >>= 1) mx = fmaxf(mx, __shfl_xor(mx, off, 32));
      float sum = 0.f;
#pragma unroll
      for (int nt = 0; nt < 8; ++nt) {
        const float p = __expf((sfr[nt][v] - mx) * SCALE);
        sfr[nt][v] = p; sum += p;
      }
#pragma unroll
      for (int off = 8; off >= 1; off >>= 1) sum += __shfl_xor(sum, off, 32);
      const float inv = 1.f / sum;
#pragma unroll
      for (int nt = 0; nt < 8; ++nt)
        Pbuf[w][mt*16 + 8*lh + v][nt*16 + ln] = f2bf(sfr[nt][v] * inv);
    }
  }

  // ---- O = P @ Vg ----
  f32x8 ofr[2][2];
#pragma unroll
  for (int i = 0; i < 2; ++i)
#pragma unroll
    for (int j = 0; j < 2; ++j) ofr[i][j] = (f32x8){0,0,0,0,0,0,0,0};

#pragma unroll
  for (int kt = 0; kt < 4; ++kt) {
    FragAB a0, a1;
    a0.u[0] = *(const u32x4*)&Pbuf[w][ln     ][kt*32 + 8*lh];
    a0.u[1] = *(const u32x4*)&Pbuf[w][ln     ][kt*32 + 16 + 8*lh];
    a1.u[0] = *(const u32x4*)&Pbuf[w][16 + ln][kt*32 + 8*lh];
    a1.u[1] = *(const u32x4*)&Pbuf[w][16 + ln][kt*32 + 16 + 8*lh];
#pragma unroll
    for (int nt = 0; nt < 2; ++nt) {
      FragAB b;
      b.u[0] = *(const u32x4*)&VT[w][nt*16 + ln][kt*32 + 8*lh];
      b.u[1] = *(const u32x4*)&VT[w][nt*16 + ln][kt*32 + 16 + 8*lh];
      ofr[0][nt] = __builtin_amdgcn_wmma_f32_16x16x32_bf16(
          false, a0.v, false, b.v, (short)0, ofr[0][nt], false, false);
      ofr[1][nt] = __builtin_amdgcn_wmma_f32_16x16x32_bf16(
          false, a1.v, false, b.v, (short)0, ofr[1][nt], false, false);
    }
  }

  // ---- store: attnb[token][h*32 + dv]  (bf16, [B*N, C] layout) ----
#pragma unroll
  for (int mt = 0; mt < 2; ++mt)
#pragma unroll
    for (int nt = 0; nt < 2; ++nt)
#pragma unroll
      for (int v = 0; v < 8; ++v) {
        const int srow = n*WIN + mt*16 + 8*lh + v;
        attnb[(tokbase + srow) * CDIM + h*DHEAD + nt*16 + ln] = f2bf(ofr[mt][nt][v]);
      }
}

// =====================================================================
extern "C" void kernel_launch(void* const* d_in, const int* in_sizes, int n_in,
                              void* d_out, int out_size, void* d_ws, size_t ws_size,
                              hipStream_t stream) {
  const float* x      = (const float*)d_in[0];
  const float* w_qkv  = (const float*)d_in[1];
  const float* b_qkv  = (const float*)d_in[2];
  const float* w_proj = (const float*)d_in[3];
  const float* b_proj = (const float*)d_in[4];
  (void)in_sizes; (void)n_in; (void)out_size; (void)ws_size;

  const size_t qkv_bytes  = (size_t)NTOK * 768 * 2;   // 96 MB bf16
  const size_t attn_bytes = (size_t)NTOK * CDIM * 2;  // 32 MB bf16
  unsigned short* qkv   = (unsigned short*)d_ws;
  unsigned short* attnb = (unsigned short*)((char*)d_ws + qkv_bytes);
  int*            gidx  = (int*)((char*)d_ws + qkv_bytes + attn_bytes);

  // 1) QKV projection: [65536,256] @ [256,768] + b -> bf16 qkv
  {
    dim3 grid(NTOK / 128, 768 / 128);
    gemm_bias_kernel<false, true><<<grid, 256, 0, stream>>>(
        (const void*)x, w_qkv, b_qkv, (void*)qkv, 768, CDIM);
  }
  // 2) Region routing: top-4 key regions per query window
  routing_kernel<<<16 * 16 * HEADS, 32, 0, stream>>>(qkv, gidx);

  // 3) Gathered windowed attention (one wave per (bt,h,window))
  attn_kernel<<<(16 * 16 * HEADS * NWIN) / 4, 128, 0, stream>>>(qkv, gidx, attnb);

  // 4) Output projection: [65536,256] @ [256,256] + b -> fp32 d_out
  {
    dim3 grid(NTOK / 128, CDIM / 128);
    gemm_bias_kernel<true, false><<<grid, 256, 0, stream>>>(
        (const void*)attnb, w_proj, b_proj, d_out, CDIM, CDIM);
  }
}
